// SCNN_43198781063249
// MI455X (gfx1250) — matrix operand order repaired and checked by reading, not compiled, not run on hardware
//
#include <hip/hip_runtime.h>
#include <hip/hip_bf16.h>
#include <stdint.h>
#include <stddef.h>

#define MDIM   4096
#define BATCH  4
#define NCOL   8     // COLORS
#define HID    240

typedef __bf16 bf16;
typedef __bf16 v16bf __attribute__((ext_vector_type(16)));
typedef __bf16 v8bf  __attribute__((ext_vector_type(8)));
typedef __bf16 v4bf  __attribute__((ext_vector_type(4)));
typedef float  v8f   __attribute__((ext_vector_type(8)));
typedef int    v4i   __attribute__((ext_vector_type(4)));
typedef int    v8i   __attribute__((ext_vector_type(8)));
typedef unsigned int v4u __attribute__((ext_vector_type(4)));

union ABfrag { v16bf v; v8bf h[2]; };

#if __has_builtin(__builtin_amdgcn_tensor_load_to_lds) && \
    __has_builtin(__builtin_amdgcn_s_wait_tensorcnt)
#define USE_TDM 1
#else
#define USE_TDM 0
#endif

#if __has_builtin(__builtin_amdgcn_global_load_async_to_lds_b128) && \
    __has_builtin(__builtin_amdgcn_s_wait_asynccnt)
#define USE_ASYNC_LDS 1
#else
#define USE_ASYNC_LDS 0
#endif

typedef __attribute__((address_space(1))) v4i* gaddr_b128;
typedef __attribute__((address_space(3))) v4i* laddr_b128;

__device__ __forceinline__ void stage16(bf16* ldst, const bf16* gsrc) {
#if USE_ASYNC_LDS
    __builtin_amdgcn_global_load_async_to_lds_b128(
        (gaddr_b128)gsrc, (laddr_b128)ldst, 0, 0);
#else
    *(v8bf*)ldst = *(const v8bf*)gsrc;
#endif
}

__device__ __forceinline__ uint32_t lds_off(const bf16* p) {
    return (uint32_t)(uintptr_t)(__attribute__((address_space(3))) const bf16*)p;
}

#if USE_TDM
// 2D TDM descriptor: tile (tile_w x tile_h) bf16 from a row-major tensor with
// line length MDIM elems; LDS padding: +4 DWORDs every 32 DWORDs stored
// (row 64 elems = 128B -> LDS row stride 144B = 72 elems = G1_LD).
__device__ __forceinline__ void tdm_load_2d(const bf16* g, const bf16* ldsp,
                                            uint32_t tensor_h, uint32_t tile_h,
                                            uint32_t tile_w) {
    const uint64_t ga = (uint64_t)g;
    v4u g0;
    g0[0] = 1u;                                     // count=1 valid descriptor
    g0[1] = lds_off(ldsp);                          // lds_addr (bytes)
    g0[2] = (uint32_t)(ga & 0xffffffffu);           // global_addr[31:0]
    g0[3] = (uint32_t)((ga >> 32) & 0x01ffffffu)    // global_addr[56:32]
          | (2u << 30);                             // type = 2 ("image")
    v8i g1;
    g1[0] = (int)((1u << 16)        // data_size = 1 -> 2-byte elements
          | (1u << 20)              // pad_enable
          | (4u << 22)              // pad_interval: 32 DWORDs (128B) stored...
          | (3u << 25));            // pad_amount:   ...then 4 DWORDs (16B) pad
    g1[1] = (int)((MDIM & 0xffffu) << 16);                 // tensor_dim0 lo16
    g1[2] = (int)((MDIM >> 16) | ((tensor_h & 0xffffu) << 16));
    g1[3] = (int)((tensor_h >> 16) | ((tile_w & 0xffffu) << 16));
    g1[4] = (int)(tile_h & 0xffffu);                       // tile_dim1 (tile_dim2=0)
    g1[5] = (int)MDIM;                                     // tensor_dim0_stride lo32
    g1[6] = 0;                                             // stride hi / dim1_stride
    g1[7] = 0;
    v4i gz4 = {};
    v8i gz8 = {};
    __builtin_amdgcn_tensor_load_to_lds(g0, g1, gz4, gz4, gz8, 0);
}
#endif

__device__ __forceinline__ void stage_wait() {
#if USE_TDM
    __builtin_amdgcn_s_wait_tensorcnt(0);
#elif USE_ASYNC_LDS
    __builtin_amdgcn_s_wait_asynccnt(0);
#endif
    __syncthreads();
}

// ---------------------------------------------------------------------------
// fp32 -> bf16 conversion (Lu and x)
// ---------------------------------------------------------------------------
__global__ void k_f32_to_bf16(const float* __restrict__ src,
                              bf16* __restrict__ dst, int n) {
    int i = (blockIdx.x * blockDim.x + threadIdx.x) * 4;
    if (i + 3 < n) {
        float4 f = *(const float4*)(src + i);
        v4bf o;
        o.x = (bf16)f.x; o.y = (bf16)f.y; o.z = (bf16)f.z; o.w = (bf16)f.w;
        *(v4bf*)(dst + i) = o;
    } else {
        for (int t = i; t < n; ++t) dst[t] = (bf16)src[t];
    }
}

// ---------------------------------------------------------------------------
// Fold theta (CoutReal, Cin, 3) fp32 into bf16 W (CoutPad x Kpad):
//   kk <  Cin : theta[o][kk][0] + theta[o][kk][1]  (W0; lower branch is [x,x])
//   kk < 2Cin : theta[o][kk-Cin][2]                (W1, applied to Lu@x)
//   else 0 (K pad); rows o >= CoutReal: 0 (Cout pad)
// ---------------------------------------------------------------------------
__global__ void k_prep_w(const float* __restrict__ theta, bf16* __restrict__ W,
                         int coutReal, int cin, int kpad, int ntot) {
    int idx = blockIdx.x * blockDim.x + threadIdx.x;
    if (idx >= ntot) return;
    int o  = idx / kpad;
    int kk = idx - o * kpad;
    float v = 0.f;
    if (o < coutReal) {
        if (kk < cin)
            v = theta[(o * cin + kk) * 3 + 0] + theta[(o * cin + kk) * 3 + 1];
        else if (kk < 2 * cin)
            v = theta[(o * cin + (kk - cin)) * 3 + 2];
    }
    W[idx] = (bf16)v;
}

// ---------------------------------------------------------------------------
// GEMM1:  Y[n][m] = sum_k Lu[m][k] * X[n][k]     (bf16 in, f32 acc, bf16 out)
// Block: 128 (m) x BN (n), 256 threads = 8 waves; wave w owns rows
// [w*16, w*16+16) x BN/16 column tiles. K-step 64, double-buffered staging
// via TDM (wave0: A-tile, wave1: B-tile) or per-lane async loads.
// One s_wait_tensorcnt/asynccnt + one barrier per 64-deep K step (8 WMMA/wave).
// ---------------------------------------------------------------------------
#define G1_LD 72   // 64 + 8 pad (matches TDM pad_interval/amount above)

template <int BN>
__device__ __forceinline__ void g1_stage(const bf16* Lu, const bf16* X,
                                         bf16* As, bf16* Bs,
                                         int m0, int n0, int k0, int tid,
                                         int ncols) {
#if USE_TDM
    const int wave = tid >> 5;
    if (wave == 0)
        tdm_load_2d(Lu + (size_t)m0 * MDIM + k0, As, MDIM, 128, 64);
    else if (wave == 1)
        tdm_load_2d(X + (size_t)n0 * MDIM + k0, Bs, (uint32_t)ncols, BN, 64);
#else
    (void)ncols;
    // A tile 128x64 = 1024 16B chunks -> 4/thread; B tile BNx64 -> BN/32 per thread
    const int row = tid >> 3;             // 0..31
    const int ko  = (tid & 7) << 3;       // 0,8,...,56
    #pragma unroll
    for (int i = 0; i < 4; ++i) {
        int r = row + i * 32;
        stage16(&As[r * G1_LD + ko], Lu + (size_t)(m0 + r) * MDIM + k0 + ko);
    }
    #pragma unroll
    for (int i = 0; i < BN / 32; ++i) {
        int r = row + i * 32;
        stage16(&Bs[r * G1_LD + ko], X + (size_t)(n0 + r) * MDIM + k0 + ko);
    }
#endif
}

template <int BN>
__global__ __launch_bounds__(256)
void k_gemm1(const bf16* __restrict__ Lu, const bf16* __restrict__ X,
             bf16* __restrict__ Y, int ncols) {
    constexpr int NT = BN / 16;
    __shared__ __align__(16) bf16 As[2][128 * G1_LD];
    __shared__ __align__(16) bf16 Bs[2][BN * G1_LD];

    const int m0   = blockIdx.x * 128;
    const int n0   = blockIdx.y * BN;
    const int tid  = threadIdx.x;
    const int wave = tid >> 5;
    const int lane = tid & 31;
    const int l16  = lane & 15;
    const int lh   = lane >> 4;

    v8f acc[NT];
    #pragma unroll
    for (int j = 0; j < NT; ++j) acc[j] = (v8f){};

    g1_stage<BN>(Lu, X, As[0], Bs[0], m0, n0, 0, tid, ncols);
    stage_wait();

    int p = 0;
    const int arow = wave * 16 + l16;
    for (int k0 = 0; k0 < MDIM; k0 += 64) {
        const bool more = (k0 + 64) < MDIM;
        if (more)  // issue next tile into the other buffer; no wait yet
            g1_stage<BN>(Lu, X, As[p ^ 1], Bs[p ^ 1], m0, n0, k0 + 64, tid, ncols);

        #pragma unroll
        for (int kh = 0; kh < 2; ++kh) {
            ABfrag a;
            const bf16* ap = &As[p][arow * G1_LD + kh * 32];
            a.h[0] = *(const v8bf*)(ap + lh * 8);
            a.h[1] = *(const v8bf*)(ap + lh * 8 + 16);
            #pragma unroll
            for (int j = 0; j < NT; ++j) {
                ABfrag b;
                const bf16* bp = &Bs[p][(j * 16 + l16) * G1_LD + kh * 32 + lh * 8];
                b.h[0] = *(const v8bf*)bp;
                b.h[1] = *(const v8bf*)(bp + 16);
                acc[j] = __builtin_amdgcn_wmma_f32_16x16x32_bf16(
                             false, a.v, false, b.v, (short)0, acc[j], false, false);
            }
        }
        if (more) { stage_wait(); p ^= 1; }
    }

    // epilogue: C layout -> Y[n][m], 8 consecutive m per lane -> b128 store
    const int mb = m0 + wave * 16 + lh * 8;
    #pragma unroll
    for (int j = 0; j < NT; ++j) {
        v8bf o8;
        #pragma unroll
        for (int r = 0; r < 8; ++r) o8[r] = (bf16)acc[j][r];
        *(v8bf*)(Y + (size_t)(n0 + j * 16 + l16) * MDIM + mb) = o8;
    }
}

// ---------------------------------------------------------------------------
// GEMM2 (channel mix): out[o][m] = sum_kk W[o][kk] * Z[kk][m],
//   Z = [act_b ; lux_b] (+K pad zeros).  Block: 16 (o) x 256 (m), 8 waves.
// A-frags straight from the small resident W; B transposed into LDS.
// ---------------------------------------------------------------------------
#define G2_LD 40
__global__ __launch_bounds__(256)
void k_gemm2(const bf16* __restrict__ act, const bf16* __restrict__ lux,
             const bf16* __restrict__ W, const float* __restrict__ bias,
             void* __restrict__ outp, int cin, int coutReal, int kpad,
             int relu, int outBf16) {
    __shared__ __align__(16) bf16 Bs[256 * G2_LD];

    const int o0   = blockIdx.x * 16;
    const int m0   = blockIdx.y * 256;
    const int bz   = blockIdx.z;
    const int tid  = threadIdx.x;
    const int wave = tid >> 5;
    const int lane = tid & 31;
    const int l16  = lane & 15;
    const int lh   = lane >> 4;
    const int twoCin = 2 * cin;

    // batch-local base rows; select between them with one cndmask per chunk
    const bf16* zr0 = act + (size_t)bz * cin * MDIM;
    const bf16* zr1 = lux + (size_t)bz * cin * MDIM - (size_t)cin * MDIM;

    v8f acc0 = {};
    v8f acc1 = {};

    for (int k0 = 0; k0 < kpad; k0 += 32) {
        // stage 32(k) x 256(m) transposed: Bs[m][k]; 1024 chunks, 4/thread
        #pragma unroll
        for (int it = 0; it < 4; ++it) {
            int c  = tid + it * 256;
            int kl = c >> 5;            // 0..31
            int ml = (c & 31) << 3;     // 0..248
            int kk = k0 + kl;
            v8bf v = {};
            if (kk < twoCin) {
                const bf16* src = (kk < cin ? zr0 : zr1) + (size_t)kk * MDIM;
                v = *(const v8bf*)(src + m0 + ml);
            }
            #pragma unroll
            for (int e = 0; e < 8; ++e) Bs[(ml + e) * G2_LD + kl] = v[e];
        }
        __syncthreads();

        ABfrag a, b0, b1;
        const bf16* wrow = W + (size_t)(o0 + l16) * kpad + k0 + lh * 8;
        a.h[0] = *(const v8bf*)wrow;
        a.h[1] = *(const v8bf*)(wrow + 16);
        int mw = wave * 32;
        b0.h[0] = *(const v8bf*)&Bs[(mw + l16) * G2_LD + lh * 8];
        b0.h[1] = *(const v8bf*)&Bs[(mw + l16) * G2_LD + lh * 8 + 16];
        b1.h[0] = *(const v8bf*)&Bs[(mw + 16 + l16) * G2_LD + lh * 8];
        b1.h[1] = *(const v8bf*)&Bs[(mw + 16 + l16) * G2_LD + lh * 8 + 16];

        acc0 = __builtin_amdgcn_wmma_f32_16x16x32_bf16(
                   false, a.v, false, b0.v, (short)0, acc0, false, false);
        acc1 = __builtin_amdgcn_wmma_f32_16x16x32_bf16(
                   false, a.v, false, b1.v, (short)0, acc1, false, false);
        __syncthreads();
    }

    // epilogue: bias + leaky-relu, write bf16 activations or fp32 output
    #pragma unroll
    for (int j = 0; j < 2; ++j) {
        v8f acc = j ? acc1 : acc0;
        int m = m0 + wave * 32 + j * 16 + l16;
        #pragma unroll
        for (int r = 0; r < 8; ++r) {
            int o = o0 + lh * 8 + r;
            if (o < coutReal) {
                float v = acc[r] + bias[o];
                if (relu) v = v > 0.f ? v : 0.01f * v;
                size_t idx = (size_t)(bz * coutReal + o) * MDIM + m;
                if (outBf16) ((bf16*)outp)[idx] = (bf16)v;
                else         ((float*)outp)[idx] = v;
            }
        }
    }
}

// ---------------------------------------------------------------------------
// Orchestration
// ---------------------------------------------------------------------------
extern "C" void kernel_launch(void* const* d_in, const int* in_sizes, int n_in,
                              void* d_out, int out_size, void* d_ws, size_t ws_size,
                              hipStream_t stream) {
    (void)in_sizes; (void)n_in; (void)out_size; (void)ws_size;
    // setup_inputs order: Ll(0) Lu(1) x(2) th1(3) b1(4) th2(5) b2(6) th3(7) b3(8)
    const float* Lu  = (const float*)d_in[1];
    const float* x   = (const float*)d_in[2];
    const float* th1 = (const float*)d_in[3];
    const float* b1  = (const float*)d_in[4];
    const float* th2 = (const float*)d_in[5];
    const float* b2  = (const float*)d_in[6];
    const float* th3 = (const float*)d_in[7];
    const float* b3  = (const float*)d_in[8];

    char* ws = (char*)d_ws;
    const size_t SZ_LU  = (size_t)MDIM * MDIM * 2;            // 32 MB
    const size_t SZ_ACT = (size_t)BATCH * HID * MDIM * 2;     // 7.5 MB
    bf16* LuB  = (bf16*)(ws);
    bf16* actA = (bf16*)(ws + SZ_LU);
    bf16* actB = (bf16*)(ws + SZ_LU + SZ_ACT);
    bf16* lux  = (bf16*)(ws + SZ_LU + 2 * SZ_ACT);
    bf16* Wb   = (bf16*)(ws + SZ_LU + 3 * SZ_ACT);

    // convert Lu and x to bf16
    {
        int n = MDIM * MDIM;
        k_f32_to_bf16<<<(n / 4 + 255) / 256, 256, 0, stream>>>(Lu, LuB, n);
        n = BATCH * NCOL * MDIM;
        k_f32_to_bf16<<<(n / 4 + 255) / 256, 256, 0, stream>>>(x, actA, n);
    }

    // ---- layer 1: Cin=8 (Kpad 32), Cout=240, ncols=32 ----
    k_prep_w<<<(240 * 32 + 255) / 256, 256, 0, stream>>>(th1, Wb, 240, NCOL, 32, 240 * 32);
    k_gemm1<32><<<dim3(MDIM / 128, (BATCH * NCOL) / 32), 256, 0, stream>>>(
        LuB, actA, lux, BATCH * NCOL);
    k_gemm2<<<dim3(240 / 16, MDIM / 256, BATCH), 256, 0, stream>>>(
        actA, lux, Wb, b1, actB, NCOL, 240, 32, 1, 1);

    // ---- layer 2: Cin=240 (Kpad 480), Cout=240, ncols=960 ----
    k_prep_w<<<(240 * 480 + 255) / 256, 256, 0, stream>>>(th2, Wb, 240, HID, 480, 240 * 480);
    k_gemm1<64><<<dim3(MDIM / 128, (BATCH * HID) / 64), 256, 0, stream>>>(
        LuB, actB, lux, BATCH * HID);
    k_gemm2<<<dim3(240 / 16, MDIM / 256, BATCH), 256, 0, stream>>>(
        actB, lux, Wb, b2, actA, HID, 240, 480, 1, 1);

    // ---- layer 3: Cin=240 (Kpad 480), Cout=8 (padded to 16), fp32 out ----
    k_prep_w<<<(16 * 480 + 255) / 256, 256, 0, stream>>>(th3, Wb, 8, HID, 480, 16 * 480);
    k_gemm1<64><<<dim3(MDIM / 128, (BATCH * HID) / 64), 256, 0, stream>>>(
        LuB, actA, lux, BATCH * HID);
    k_gemm2<<<dim3(1, MDIM / 256, BATCH), 256, 0, stream>>>(
        actA, lux, Wb, b3, d_out, HID, 8, 480, 0, 0);
}
